// MAB_51307679318304
// MI455X (gfx1250) — matrix operand order repaired and checked by reading, not compiled, not run on hardware
//
#include <hip/hip_runtime.h>
#include <hip/hip_bf16.h>

// MI455X / gfx1250: wave32, WMMA f32_16x16x32_f16, f32 accumulate.
// Transformer cross-attention block, flash-softmax, all GEMMs on the WMMA pipe.
// v2: register double-buffered GEMM (partial loadcnt waits instead of full
// drains), software-pipelined attention (K/V fragment loads hidden under
// softmax VALU), 8-wave GEMM blocks.

#define B_    4
#define LSEQ  2048
#define EMB   256
#define NHEAD 4
#define HDIM  64
#define KDIM  256
#define LN_EPS 1e-5f

typedef __attribute__((ext_vector_type(16))) _Float16 v16h;
typedef __attribute__((ext_vector_type(8)))  _Float16 v8h;
typedef __attribute__((ext_vector_type(8)))  float    v8f;

__device__ __forceinline__ v16h cat8(v8h lo, v8h hi) {
  return __builtin_shufflevector(lo, hi, 0,1,2,3,4,5,6,7,8,9,10,11,12,13,14,15);
}
// B-fragment: 16 contiguous halfs (lane = column N, K = base + sel*16 + i)
__device__ __forceinline__ v16h load_bfrag(const _Float16* p) {
  v8h lo = *(const v8h*)(p);
  v8h hi = *(const v8h*)(p + 8);
  return cat8(lo, hi);
}
// A-fragment: lane = row M; K = sel*16 + i
__device__ __forceinline__ v16h load_afrag(const _Float16* arow, int kk, int sel) {
  v8h lo = *(const v8h*)(arow + kk + sel * 16);
  v8h hi = *(const v8h*)(arow + kk + sel * 16 + 8);
  return cat8(lo, hi);
}
__device__ __forceinline__ v8f wmma16(v16h a, v16h b, v8f c) {
  return __builtin_amdgcn_wmma_f32_16x16x32_f16(false, a, false, b, (short)0, c,
                                                false, false);
}

// ---------------------------------------------------------------- converts
__global__ void cvt_f32_f16(const float* __restrict__ in,
                            _Float16* __restrict__ out, int n) {
  int i = blockIdx.x * blockDim.x + threadIdx.x;
  if (i < n) out[i] = (_Float16)in[i];
}

// V: [B,L,E] (head-major cols) -> Vt: [B,H,D,L] so PV B-frags are contiguous
__global__ void transpose_v(const _Float16* __restrict__ v,
                            _Float16* __restrict__ vt) {
  size_t i = (size_t)blockIdx.x * blockDim.x + threadIdx.x;
  int key = (int)(i & (LSEQ - 1));
  size_t j = i >> 11;
  int d = (int)(j & (HDIM - 1));
  j >>= 6;
  int h = (int)(j & (NHEAD - 1));
  int b = (int)(j >> 2);
  vt[i] = v[((size_t)b * LSEQ + key) * EMB + h * HDIM + d];
}

// ---------------------------------------------------------------- WMMA GEMM
// C[M,N] = A16[M,K=256] * W16[N,K=256]^T + bias ; optional relu / f32 / f16 out
// block = 256 threads (8 waves along M), wave computes 16x64, macro 128x64.
// Register double-buffered: frags for k+32 are in flight during WMMAs for k.
__global__ void gemm_wmma_256(const _Float16* __restrict__ A,
                              const _Float16* __restrict__ W,
                              const float* __restrict__ bias,
                              float* __restrict__ outF,
                              _Float16* __restrict__ outH,
                              int M, int N, int relu) {
  const int wid = threadIdx.x >> 5;
  const int l   = threadIdx.x & 31;
  const int sel = l >> 4;        // lane half selects K sub-block
  const int lan = l & 15;        // A: row M ; B: column N ; C: column N
  const int m0 = blockIdx.x * 128 + wid * 16;
  const int n0 = blockIdx.y * 64;

  const _Float16* arow = A + (size_t)(m0 + lan) * KDIM;
  const _Float16* wrow[4];
#pragma unroll
  for (int t = 0; t < 4; ++t)
    wrow[t] = W + (size_t)(n0 + t * 16 + lan) * KDIM + sel * 16;

  v8f acc[4] = {};
  v16h af = load_afrag(arow, 0, sel);
  v16h bf[4];
#pragma unroll
  for (int t = 0; t < 4; ++t) bf[t] = load_bfrag(wrow[t]);

#pragma unroll
  for (int kk = 0; kk < KDIM; kk += 32) {
    v16h afn;
    v16h bfn[4];
    if (kk + 32 < KDIM) {                      // prefetch next K-step frags
      __builtin_prefetch(arow + kk + 96, 0, 1);
      afn = load_afrag(arow, kk + 32, sel);
#pragma unroll
      for (int t = 0; t < 4; ++t) bfn[t] = load_bfrag(wrow[t] + kk + 32);
    }
#pragma unroll
    for (int t = 0; t < 4; ++t) acc[t] = wmma16(af, bf[t], acc[t]);
    af = afn;
#pragma unroll
    for (int t = 0; t < 4; ++t) bf[t] = bfn[t];
  }

#pragma unroll
  for (int t = 0; t < 4; ++t) {
    int n = n0 + t * 16 + lan;
    float bv = bias[n];
#pragma unroll
    for (int r = 0; r < 8; ++r) {
      int m = m0 + r + 8 * sel;                // C layout: VGPR r, lane half
      float v = acc[t][r] + bv;
      if (relu) v = v > 0.f ? v : 0.f;
      if (outF) outF[(size_t)m * N + n] = v;
      if (outH) outH[(size_t)m * N + n] = (_Float16)v;
    }
  }
}

// ---------------------------------------------------------------- attention
// One wave per (b, head, 16-query tile). Flash softmax over 32-key blocks.
// Pipelined: V frags for block j and K frags for block j+32 are loaded before
// the softmax VALU work, which hides their latency (WMMA/VALU co-execution).
__global__ void attn_flash(const _Float16* __restrict__ q,
                           const _Float16* __restrict__ k,
                           const _Float16* __restrict__ vt,
                           _Float16* __restrict__ o) {
  const int qt = blockIdx.x, h = blockIdx.y, b = blockIdx.z;
  const int l = threadIdx.x;
  const int sel = l >> 4, lan = l & 15;
  __shared__ __align__(32) _Float16 pbuf[16 * 32];

  // Q A-fragments for d = 0..31 and 32..63 (held for the whole row)
  const _Float16* qrow = q + ((size_t)b * LSEQ + (size_t)qt * 16 + lan) * EMB + h * HDIM;
  v16h qa0 = cat8(*(const v8h*)(qrow + sel * 16),      *(const v8h*)(qrow + 8 + sel * 16));
  v16h qa1 = cat8(*(const v8h*)(qrow + 32 + sel * 16), *(const v8h*)(qrow + 40 + sel * 16));

  const _Float16* kb = k + (size_t)b * LSEQ * EMB + h * HDIM;
  const _Float16* vb = vt + ((size_t)(b * NHEAD + h)) * HDIM * LSEQ;

  v8f acc[4] = {};
  float rmax[8], rsum[8];
#pragma unroll
  for (int r = 0; r < 8; ++r) { rmax[r] = -3.4e38f; rsum[r] = 0.f; }

  // preload K B-fragments for the first key block
  v16h kf[2][2];
#pragma unroll
  for (int t = 0; t < 2; ++t) {
    const _Float16* kr = kb + (size_t)(t * 16 + lan) * EMB;
    kf[t][0] = load_bfrag(kr + sel * 16);
    kf[t][1] = load_bfrag(kr + 32 + sel * 16);
  }

  for (int j = 0; j < LSEQ; j += 32) {
    v8f s[2];
#pragma unroll
    for (int t = 0; t < 2; ++t) {             // S = Q * K^T, 16x16 per tile
      v8f z = {};
      z = wmma16(qa0, kf[t][0], z);
      z = wmma16(qa1, kf[t][1], z);
      s[t] = z;
    }
    // issue V frags (block j) and K frags (block j+32) under the softmax
    v16h vf[4];
#pragma unroll
    for (int t = 0; t < 4; ++t)
      vf[t] = load_bfrag(vb + (size_t)(t * 16 + lan) * LSEQ + j + sel * 16);
    if (j + 32 < LSEQ) {
#pragma unroll
      for (int t = 0; t < 2; ++t) {
        const _Float16* kr = kb + (size_t)(j + 32 + t * 16 + lan) * EMB;
        kf[t][0] = load_bfrag(kr + sel * 16);
        kf[t][1] = load_bfrag(kr + 32 + sel * 16);
      }
    }
    // online softmax: row stats live in 16-lane halves (C layout rows)
#pragma unroll
    for (int r = 0; r < 8; ++r) {
      s[0][r] *= 0.125f;                      // 1/sqrt(64)
      s[1][r] *= 0.125f;
      float m = fmaxf(s[0][r], s[1][r]);
      m = fmaxf(m, __shfl_xor(m, 1, 32));
      m = fmaxf(m, __shfl_xor(m, 2, 32));
      m = fmaxf(m, __shfl_xor(m, 4, 32));
      m = fmaxf(m, __shfl_xor(m, 8, 32));
      float nm = fmaxf(rmax[r], m);
      float sc = __expf(rmax[r] - nm);
      rmax[r] = nm;
      float p0 = __expf(s[0][r] - nm);
      float p1 = __expf(s[1][r] - nm);
      float ts = p0 + p1;
      ts += __shfl_xor(ts, 1, 32);
      ts += __shfl_xor(ts, 2, 32);
      ts += __shfl_xor(ts, 4, 32);
      ts += __shfl_xor(ts, 8, 32);
      rsum[r] = rsum[r] * sc + ts;
#pragma unroll
      for (int t2 = 0; t2 < 4; ++t2) acc[t2][r] *= sc;
      int mr = r + 8 * sel;                   // reshape C-layout -> row-major
      pbuf[mr * 32 + lan] = (_Float16)p0;
      pbuf[mr * 32 + 16 + lan] = (_Float16)p1;
    }
    asm volatile("s_wait_dscnt 0" ::: "memory");  // LDS RAW before re-read
    const _Float16* prow = pbuf + lan * 32;       // P as A-fragment (K=keys)
    v16h pa = cat8(*(const v8h*)(prow + sel * 16),
                   *(const v8h*)(prow + 8 + sel * 16));
#pragma unroll
    for (int t = 0; t < 4; ++t)               // O += P * V (d tile t)
      acc[t] = wmma16(pa, vf[t], acc[t]);
    asm volatile("s_wait_dscnt 0" ::: "memory");  // reads drain before rewrite
  }
#pragma unroll
  for (int t = 0; t < 4; ++t) {
#pragma unroll
    for (int r = 0; r < 8; ++r) {
      int m = qt * 16 + r + 8 * sel;
      float v = acc[t][r] / rsum[r];
      o[((size_t)b * LSEQ + m) * EMB + h * HDIM + t * 16 + lan] = (_Float16)v;
    }
  }
}

// ---------------------------------------------------------------- LayerNorm
// out = LN(a + b) * gamma + beta ; wave per row, 4 rows per block.
__global__ void ln_residual(const float* __restrict__ a,
                            const float* __restrict__ bres,
                            const float* __restrict__ gamma,
                            const float* __restrict__ beta,
                            float* __restrict__ outF,
                            _Float16* __restrict__ outH) {
  int row = blockIdx.x * 4 + (threadIdx.x >> 5);
  int l = threadIdx.x & 31;
  const float* pa = a + (size_t)row * EMB;
  const float* pb = bres + (size_t)row * EMB;
  float v[8];
  float s = 0.f;
#pragma unroll
  for (int i = 0; i < 8; ++i) {
    int idx = l + i * 32;
    v[i] = pa[idx] + pb[idx];
    s += v[i];
  }
  for (int mask = 16; mask; mask >>= 1) s += __shfl_xor(s, mask, 32);
  float mean = s * (1.f / EMB);
  float qv = 0.f;
#pragma unroll
  for (int i = 0; i < 8; ++i) { float d = v[i] - mean; qv += d * d; }
  for (int mask = 16; mask; mask >>= 1) qv += __shfl_xor(qv, mask, 32);
  float rstd = rsqrtf(qv * (1.f / EMB) + LN_EPS);
#pragma unroll
  for (int i = 0; i < 8; ++i) {
    int idx = l + i * 32;
    float o = (v[i] - mean) * rstd * gamma[idx] + beta[idx];
    if (outF) outF[(size_t)row * EMB + idx] = o;
    if (outH) outH[(size_t)row * EMB + idx] = (_Float16)o;
  }
}

// ---------------------------------------------------------------- launcher
extern "C" void kernel_launch(void* const* d_in, const int* in_sizes, int n_in,
                              void* d_out, int out_size, void* d_ws,
                              size_t ws_size, hipStream_t stream) {
  (void)in_sizes; (void)n_in; (void)out_size; (void)ws_size;
  const float* x   = (const float*)d_in[0];
  const float* y   = (const float*)d_in[1];
  const float* wq  = (const float*)d_in[2];
  const float* bq  = (const float*)d_in[3];
  const float* wkv = (const float*)d_in[4];
  const float* bkv = (const float*)d_in[5];
  const float* wo  = (const float*)d_in[6];
  const float* bo  = (const float*)d_in[7];
  const float* w1  = (const float*)d_in[8];
  const float* b1  = (const float*)d_in[9];
  const float* w2  = (const float*)d_in[10];
  const float* b2  = (const float*)d_in[11];
  const float* g0  = (const float*)d_in[12];
  const float* be0 = (const float*)d_in[13];
  const float* g1  = (const float*)d_in[14];
  const float* be1 = (const float*)d_in[15];
  float* out = (float*)d_out;

  char* ws = (char*)d_ws;
  const size_t M = (size_t)B_ * LSEQ;                   // 8192 rows
  const size_t SZH = M * EMB * sizeof(_Float16);        // 4 MiB per f16 act
  // lifetime-overlaid workspace map (~29 MiB total):
  _Float16* x16  = (_Float16*)(ws + 0 * SZH);
  _Float16* y16  = (_Float16*)(ws + 1 * SZH);
  _Float16* q16  = (_Float16*)(ws + 2 * SZH);
  _Float16* k16  = (_Float16*)(ws + 3 * SZH);
  _Float16* v16  = (_Float16*)(ws + 4 * SZH);
  _Float16* vt16 = (_Float16*)(ws + 5 * SZH);
  _Float16* o16  = (_Float16*)(ws + 6 * SZH);
  float*    t32  = (float*)(ws + 0 * SZH);   // reuses x16+y16 (dead after qkv)
  float*    h32  = (float*)(ws + 2 * SZH);   // reuses q16+k16 (dead after attn)
  _Float16* h16  = (_Float16*)(ws + 4 * SZH);// reuses v16 (dead after transpose)
  _Float16* f116 = (_Float16*)(ws + 5 * SZH);// reuses vt16 (dead after attn)
  float*    t232 = (float*)(ws + 0 * SZH);   // reuses t32 (dead after ln1)
  char* wb = ws + 7 * SZH;
  _Float16* wq16  = (_Float16*)(wb);
  _Float16* wkv16 = (_Float16*)(wb + 131072);
  _Float16* wo16  = (_Float16*)(wb + 393216);
  _Float16* w116  = (_Float16*)(wb + 524288);
  _Float16* w216  = (_Float16*)(wb + 655360);

  auto cvt = [&](const float* s, _Float16* d, int n) {
    cvt_f32_f16<<<dim3((n + 255) / 256), dim3(256), 0, stream>>>(s, d, n);
  };
  cvt(x, x16, (int)(M * EMB));
  cvt(y, y16, (int)(M * EMB));
  cvt(wq, wq16, EMB * EMB);
  cvt(wkv, wkv16, 2 * EMB * EMB);
  cvt(wo, wo16, EMB * EMB);
  cvt(w1, w116, EMB * EMB);
  cvt(w2, w216, EMB * EMB);

  dim3 ggrid((unsigned)(M / 128), EMB / 64), gblk(256);
  // q = x wq^T + bq ; k/v from wkv (rows [0,E) and [E,2E))
  gemm_wmma_256<<<ggrid, gblk, 0, stream>>>(x16, wq16, bq, nullptr, q16, (int)M, EMB, 0);
  gemm_wmma_256<<<ggrid, gblk, 0, stream>>>(y16, wkv16, bkv, nullptr, k16, (int)M, EMB, 0);
  gemm_wmma_256<<<ggrid, gblk, 0, stream>>>(y16, wkv16 + EMB * EMB, bkv + EMB,
                                            nullptr, v16, (int)M, EMB, 0);

  transpose_v<<<dim3((B_ * NHEAD * HDIM * LSEQ) / 256), dim3(256), 0, stream>>>(v16, vt16);

  attn_flash<<<dim3(LSEQ / 16, NHEAD, B_), dim3(32), 0, stream>>>(q16, k16, vt16, o16);

  // out-proj -> t32 ; h = LN(x + t32)
  gemm_wmma_256<<<ggrid, gblk, 0, stream>>>(o16, wo16, bo, t32, nullptr, (int)M, EMB, 0);
  ln_residual<<<dim3((unsigned)(M / 4)), dim3(128), 0, stream>>>(x, t32, g0, be0, h32, h16);

  // FFN: relu(h w1^T + b1) w2^T + b2 ; out = LN(h + ffn)
  gemm_wmma_256<<<ggrid, gblk, 0, stream>>>(h16, w116, b1, nullptr, f116, (int)M, EMB, 1);
  gemm_wmma_256<<<ggrid, gblk, 0, stream>>>(f116, w216, b2, t232, nullptr, (int)M, EMB, 0);
  ln_residual<<<dim3((unsigned)(M / 4)), dim3(128), 0, stream>>>(h32, t232, g1, be1, out, nullptr);
}
